// Model_78168404787399
// MI455X (gfx1250) — matrix operand order repaired and checked by reading, not compiled, not run on hardware
//
#include <hip/hip_runtime.h>
#include <hip/hip_bf16.h>

// ---------------------------------------------------------------------------
// Show-Attend-Tell decoder on MI455X (gfx1250, wave32, WMMA).
// GEMMs run on v_wmma_f32_16x16x32_bf16 using bf16x3 split precision
// (hi/lo decomposition of both operands, fp32 accumulate) for near-fp32
// accuracy at bf16 matrix-unit throughput.
//
// Round-1 fixes:
//  * NO divergent control flow around WMMA (ISA: EXEC must be all-1s and
//    WMMA is not skipped on EXEC==0). Edge N-tiles are handled by clamping
//    B-row bases (loads stay in bounds) and predicating only the stores.
//  * K-loop software pipelined with explicit double buffering (K/32 is even
//    at every call site) so fragment loads overlap the 12 WMMAs per step.
// ---------------------------------------------------------------------------

typedef __bf16 bf16_t;
typedef __attribute__((ext_vector_type(16))) bf16_t v16bf;
typedef __attribute__((ext_vector_type(8)))  bf16_t v8bf;
typedef __attribute__((ext_vector_type(8)))  float  v8f;

#define B_   64
#define P_   196
#define T_   20
#define H_   512
#define A_   512
#define E_   2048
#define V_   10000
#define EMB_ 300
#define KI_  2348    // E + EMB
#define KIP_ 2368    // KI padded to multiple of 32
#define G4H_ 2048    // 4*H

// ---------------------------------------------------------------------------
// WMMA fragment gathers (per CDNA5 ISA 7.12.2 16-bit layouts).
// A fragment 16x32 (MxK): lane = half*16 + r; row m = r.
//   half=0: VGPR0-3 K=0..7,  VGPR4-7 K=16..23
//   half=1: VGPR0-3 K=8..15, VGPR4-7 K=24..31
// B fragment 32x16 (KxN), stored as BT[N][K] row-major: lane n = r.
//   same K order as A: half=0: K=0..7,16..23; half=1: K=8..15,24..31.
// ---------------------------------------------------------------------------
__device__ inline v16bf frag_a(const bf16_t* __restrict__ base, int lda,
                               int row, int k0, int half) {
  const bf16_t* p = base + (size_t)row * lda + k0 + (half << 3);
  v8bf x0 = *(const v8bf*)p;
  v8bf x1 = *(const v8bf*)(p + 16);
  v16bf f;
#pragma unroll
  for (int i = 0; i < 8; ++i) { f[i] = x0[i]; f[8 + i] = x1[i]; }
  return f;
}

__device__ inline v16bf frag_b(const bf16_t* __restrict__ base, int ldb,
                               int row, int k0, int half) {
  const bf16_t* p = base + (size_t)row * ldb + k0 + (half << 3);
  v8bf x0 = *(const v8bf*)p;
  v8bf x1 = *(const v8bf*)(p + 16);
  v16bf f;
#pragma unroll
  for (int i = 0; i < 8; ++i) { f[i] = x0[i]; f[8 + i] = x1[i]; }
  return f;
}

struct Frags { v16bf ah[2], al[2], bh[2], bl[2]; };

__device__ inline void load_frags(Frags& F,
                                  const bf16_t* __restrict__ Ahi,
                                  const bf16_t* __restrict__ Alo, int lda,
                                  const bf16_t* __restrict__ Bhi,
                                  const bf16_t* __restrict__ Blo, int ldb,
                                  int m0, int nb0, int nb1, int r, int half,
                                  int k0) {
#pragma unroll
  for (int i = 0; i < 2; ++i) {
    const int row = m0 + i * 16 + r;
    F.ah[i] = frag_a(Ahi, lda, row, k0, half);
    F.al[i] = frag_a(Alo, lda, row, k0, half);
  }
  F.bh[0] = frag_b(Bhi, ldb, nb0 + r, k0, half);
  F.bl[0] = frag_b(Blo, ldb, nb0 + r, k0, half);
  F.bh[1] = frag_b(Bhi, ldb, nb1 + r, k0, half);
  F.bl[1] = frag_b(Blo, ldb, nb1 + r, k0, half);
}

__device__ inline void mma_step(v8f acc[2][2], const Frags& F) {
#pragma unroll
  for (int i = 0; i < 2; ++i)
#pragma unroll
    for (int j = 0; j < 2; ++j) {
      acc[i][j] = __builtin_amdgcn_wmma_f32_16x16x32_bf16(
          false, F.al[i], false, F.bh[j], (short)0, acc[i][j], false, false);
      acc[i][j] = __builtin_amdgcn_wmma_f32_16x16x32_bf16(
          false, F.ah[i], false, F.bl[j], (short)0, acc[i][j], false, false);
      acc[i][j] = __builtin_amdgcn_wmma_f32_16x16x32_bf16(
          false, F.ah[i], false, F.bh[j], (short)0, acc[i][j], false, false);
    }
}

// ---------------------------------------------------------------------------
// C[M,N] = A[M,K] @ B[K,N] (+bias) (+=C if accIn), bf16x3 via WMMA.
// A planes: row-major MxK (lda). B planes: transposed, row-major NxK (ldb).
// Block: 256 threads = 8 waves as 2(M) x 4(N); wave tile 32x32 (2x2 WMMA).
// Requires M % 64 == 0, N % 16 == 0, N >= 32, K % 64 == 0 (all call sites).
// ---------------------------------------------------------------------------
__global__ __launch_bounds__(256) void k_gemm(
    const bf16_t* __restrict__ Ahi, const bf16_t* __restrict__ Alo, int lda,
    const bf16_t* __restrict__ Bhi, const bf16_t* __restrict__ Blo, int ldb,
    const float* __restrict__ bias, float* __restrict__ C, long ldc,
    int M, int N, int K, int accIn)
{
  const int lane = threadIdx.x & 31;
  const int wave = threadIdx.x >> 5;
  const int wm = wave >> 2;                   // 0..1
  const int wn = wave & 3;                    // 0..3
  const int m0 = blockIdx.x * 64  + wm * 32;  // always < M (M % 64 == 0)
  const int n0 = blockIdx.y * 128 + wn * 32;  // may exceed N on edge blocks
  const int half = lane >> 4;
  const int r    = lane & 15;
  // Clamped B-tile row bases: loads always in-bounds; out-of-range tiles
  // compute garbage that is never stored. Keeps EXEC all-1s for every WMMA.
  const int nb0 = (n0      < N) ? n0        : (N - 16);
  const int nb1 = (n0 + 16 < N) ? (n0 + 16) : (N - 16);

  v8f zero = {};
  v8f acc[2][2];
#pragma unroll
  for (int i = 0; i < 2; ++i)
#pragma unroll
    for (int j = 0; j < 2; ++j) acc[i][j] = zero;

  // Double-buffered K loop (K/32 even). All branches are on the scalar K,
  // so EXEC is untouched around the WMMAs.
  Frags f0, f1;
  load_frags(f0, Ahi, Alo, lda, Bhi, Blo, ldb, m0, nb0, nb1, r, half, 0);
  for (int k0 = 0; k0 < K; k0 += 64) {
    load_frags(f1, Ahi, Alo, lda, Bhi, Blo, ldb, m0, nb0, nb1, r, half,
               k0 + 32);
    mma_step(acc, f0);
    int kx = k0 + 64;
    kx = (kx < K) ? kx : (K - 32);            // harmless reload on last iter
    load_frags(f0, Ahi, Alo, lda, Bhi, Blo, ldb, m0, nb0, nb1, r, half, kx);
    mma_step(acc, f1);
  }

  // Epilogue. C/D layout: lane holds column n0+j*16+r; VGPR reg holds row
  // reg + 8*half. Store predication (divergent only on the N edge) is legal:
  // no WMMA follows.
#pragma unroll
  for (int i = 0; i < 2; ++i)
#pragma unroll
    for (int j = 0; j < 2; ++j) {
      const int nt = n0 + j * 16;
      if (nt < N) {
        const int n = nt + r;
        const float bv = bias ? bias[n] : 0.f;
        float* cp = C + (long)(m0 + i * 16 + half * 8) * ldc + n;
#pragma unroll
        for (int reg = 0; reg < 8; ++reg) {
          float v = acc[i][j][reg] + bv;
          if (accIn) v += *cp;
          *cp = v;
          cp += ldc;
        }
      }
    }
}

// ---------------------------------------------------------------------------
// Weight prep: W[K][N] fp32 -> WT_hi/lo[N][Kp] bf16 (zero-padded K).
// ---------------------------------------------------------------------------
__global__ void k_transpose_split(const float* __restrict__ W,
                                  bf16_t* __restrict__ Thi,
                                  bf16_t* __restrict__ Tlo,
                                  int K, int N, int Kp)
{
  int idx = blockIdx.x * blockDim.x + threadIdx.x;
  if (idx >= N * Kp) return;
  int n = idx / Kp, k = idx - n * Kp;
  float v = (k < K) ? W[(size_t)k * N + n] : 0.f;
  bf16_t h = (bf16_t)v;
  Thi[idx] = h;
  Tlo[idx] = (bf16_t)(v - (float)h);
}

// fp32 -> bf16 hi/lo planes (same layout)
__global__ void k_split(const float* __restrict__ in,
                        bf16_t* __restrict__ hi, bf16_t* __restrict__ lo,
                        long n)
{
  long i = (long)blockIdx.x * blockDim.x + threadIdx.x;
  if (i >= n) return;
  float v = in[i];
  bf16_t h = (bf16_t)v;
  hi[i] = h;
  lo[i] = (bf16_t)(v - (float)h);
}

// mean over P of x[B,P,E] -> mean fp32 + hi/lo planes
__global__ void k_mean(const float* __restrict__ x, float* __restrict__ mean,
                       bf16_t* __restrict__ mhi, bf16_t* __restrict__ mlo)
{
  int e = blockIdx.x * blockDim.x + threadIdx.x;
  int b = blockIdx.y;
  if (e >= E_) return;
  float s = 0.f;
  for (int p = 0; p < P_; ++p) s += x[((size_t)b * P_ + p) * E_ + e];
  s *= (1.f / P_);
  int idx = b * E_ + e;
  mean[idx] = s;
  bf16_t h = (bf16_t)s;
  mhi[idx] = h;
  mlo[idx] = (bf16_t)(s - (float)h);
}

// logits[b,p] = relu(attn1[b,p,:] + attn2[b,:]) . Wfa + bfa  (one wave/row)
__global__ void k_attn_logits(const float* __restrict__ attn1,
                              const float* __restrict__ attn2,
                              const float* __restrict__ Wfa,
                              const float* __restrict__ bfa,
                              float* __restrict__ logits)
{
  int wave = threadIdx.x >> 5, lane = threadIdx.x & 31;
  int row = blockIdx.x * 8 + wave;           // row in [0, B*P)
  if (row >= B_ * P_) return;
  int b = row / P_;
  const float* a1 = attn1 + (size_t)row * A_;
  const float* a2 = attn2 + (size_t)b * A_;
  float s = 0.f;
  for (int a = lane; a < A_; a += 32) {
    float v = a1[a] + a2[a];
    v = fmaxf(v, 0.f);
    s += v * Wfa[a];
  }
#pragma unroll
  for (int o = 16; o; o >>= 1) s += __shfl_xor(s, o, 32);
  if (lane == 0) logits[row] = s + bfa[0];
}

// softmax over P per batch row; writes step alpha and the alphas output slice
__global__ void k_softmax(const float* __restrict__ logits,
                          float* __restrict__ alpha,
                          float* __restrict__ alphas_out, int t)
{
  int b = blockIdx.x;
  int tid = threadIdx.x, lane = tid & 31, wave = tid >> 5;
  __shared__ float red[8];
  float v = (tid < P_) ? logits[b * P_ + tid] : -3.4e38f;
  float m = v;
#pragma unroll
  for (int o = 16; o; o >>= 1) m = fmaxf(m, __shfl_xor(m, o, 32));
  if (lane == 0) red[wave] = m;
  __syncthreads();
  float mm = red[0];
#pragma unroll
  for (int i = 1; i < 8; ++i) mm = fmaxf(mm, red[i]);
  __syncthreads();
  float e = (tid < P_) ? __expf(v - mm) : 0.f;
  float s = e;
#pragma unroll
  for (int o = 16; o; o >>= 1) s += __shfl_xor(s, o, 32);
  if (lane == 0) red[wave] = s;
  __syncthreads();
  float ss = 0.f;
#pragma unroll
  for (int i = 0; i < 8; ++i) ss += red[i];
  float a = e / ss;
  if (tid < P_) {
    alpha[b * P_ + tid] = a;
    alphas_out[((size_t)b * T_ + t) * P_ + tid] = a;
  }
}

// awe[b,e] = sum_p alpha[b,p] * x[b,p,e]
__global__ void k_awe(const float* __restrict__ x,
                      const float* __restrict__ alpha,
                      float* __restrict__ awe)
{
  int e = blockIdx.x * blockDim.x + threadIdx.x;
  int b = blockIdx.y;
  __shared__ float sal[P_];
  if (threadIdx.x < P_) sal[threadIdx.x] = alpha[b * P_ + threadIdx.x];
  __syncthreads();
  if (e >= E_) return;
  float s = 0.f;
  for (int p = 0; p < P_; ++p) s += sal[p] * x[((size_t)b * P_ + p) * E_ + e];
  awe[b * E_ + e] = s;
}

// inp[b, 0:300]=emb[cap[b,t]], inp[b,300:2348]=sigmoid(gateLin)*awe, pad zeros
__global__ void k_build_inp(const int* __restrict__ captions,
                            const float* __restrict__ emb,
                            const float* __restrict__ gateLin,
                            const float* __restrict__ awe,
                            bf16_t* __restrict__ inp_hi,
                            bf16_t* __restrict__ inp_lo, int t)
{
  int idx = blockIdx.x * blockDim.x + threadIdx.x;
  if (idx >= B_ * KIP_) return;
  int b = idx / KIP_, col = idx - b * KIP_;
  float v;
  if (col < EMB_) {
    int tok = captions[b * T_ + t];
    v = emb[(size_t)tok * EMB_ + col];
  } else if (col < KI_) {
    int e = col - EMB_;
    float g = gateLin[b * E_ + e];
    g = 1.f / (1.f + __expf(-g));
    v = g * awe[b * E_ + e];
  } else {
    v = 0.f;
  }
  bf16_t h = (bf16_t)v;
  inp_hi[idx] = h;
  inp_lo[idx] = (bf16_t)(v - (float)h);
}

// LSTM pointwise; updates h,c and refreshes h's bf16 hi/lo planes
__global__ void k_lstm(const float* __restrict__ gates,
                       float* __restrict__ h, float* __restrict__ c,
                       bf16_t* __restrict__ h_hi, bf16_t* __restrict__ h_lo)
{
  int idx = blockIdx.x * blockDim.x + threadIdx.x;
  if (idx >= B_ * H_) return;
  int b = idx / H_, j = idx - b * H_;
  const float* g = gates + (size_t)b * G4H_;
  float i_ = 1.f / (1.f + __expf(-g[j]));
  float f_ = 1.f / (1.f + __expf(-g[H_ + j]));
  float gg = tanhf(g[2 * H_ + j]);
  float o_ = 1.f / (1.f + __expf(-g[3 * H_ + j]));
  float cn = f_ * c[idx] + i_ * gg;
  float hn = o_ * tanhf(cn);
  c[idx] = cn;
  h[idx] = hn;
  bf16_t hb = (bf16_t)hn;
  h_hi[idx] = hb;
  h_lo[idx] = (bf16_t)(hn - (float)hb);
}

// ---------------------------------------------------------------------------
// Host side
// ---------------------------------------------------------------------------
static inline void gemm(hipStream_t s,
                        const bf16_t* Ahi, const bf16_t* Alo, int lda,
                        const bf16_t* Bhi, const bf16_t* Blo, int ldb,
                        const float* bias, float* C, long ldc,
                        int M, int N, int K, int accIn)
{
  dim3 g((M + 63) / 64, (N + 127) / 128), b(256);
  k_gemm<<<g, b, 0, s>>>(Ahi, Alo, lda, Bhi, Blo, ldb, bias, C, ldc, M, N, K, accIn);
}

extern "C" void kernel_launch(void* const* d_in, const int* in_sizes, int n_in,
                              void* d_out, int out_size, void* d_ws, size_t ws_size,
                              hipStream_t stream)
{
  const float* x    = (const float*)d_in[0];
  const int*   caps = (const int*)  d_in[1];
  const float* We_a = (const float*)d_in[3];
  const float* be_a = (const float*)d_in[4];
  const float* Wd_a = (const float*)d_in[5];
  const float* bd_a = (const float*)d_in[6];
  const float* Wf_a = (const float*)d_in[7];
  const float* bf_a = (const float*)d_in[8];
  const float* emb  = (const float*)d_in[9];
  const float* Wih  = (const float*)d_in[10];
  const float* bih  = (const float*)d_in[11];
  const float* Whh  = (const float*)d_in[12];
  const float* bhh  = (const float*)d_in[13];
  const float* Wh0  = (const float*)d_in[14];
  const float* bh0  = (const float*)d_in[15];
  const float* Wc0  = (const float*)d_in[16];
  const float* bc0  = (const float*)d_in[17];
  const float* Wfb  = (const float*)d_in[18];
  const float* bfb  = (const float*)d_in[19];
  const float* Wfc  = (const float*)d_in[20];
  const float* bfc  = (const float*)d_in[21];

  float* preds_out  = (float*)d_out;                      // [B,T,V]
  float* alphas_out = preds_out + (size_t)B_ * T_ * V_;   // [B,T,P]

  // ---- bump allocator over d_ws (deterministic, same layout every call) ----
  size_t off = 0;
  auto alloc = [&](size_t bytes) -> void* {
    void* p = (char*)d_ws + off;
    off += (bytes + 255) & ~(size_t)255;
    return p;
  };
  const size_t MP = (size_t)B_ * P_;                 // 12544 rows
  bf16_t* Xhi   = (bf16_t*)alloc(MP * E_ * 2);
  bf16_t* Xlo   = (bf16_t*)alloc(MP * E_ * 2);
  float*  attn1 = (float*) alloc(MP * A_ * 4);
  bf16_t* WeaThi = (bf16_t*)alloc((size_t)A_ * E_ * 2);
  bf16_t* WeaTlo = (bf16_t*)alloc((size_t)A_ * E_ * 2);
  bf16_t* WdaThi = (bf16_t*)alloc((size_t)A_ * H_ * 2);
  bf16_t* WdaTlo = (bf16_t*)alloc((size_t)A_ * H_ * 2);
  bf16_t* Wh0Thi = (bf16_t*)alloc((size_t)H_ * E_ * 2);
  bf16_t* Wh0Tlo = (bf16_t*)alloc((size_t)H_ * E_ * 2);
  bf16_t* Wc0Thi = (bf16_t*)alloc((size_t)H_ * E_ * 2);
  bf16_t* Wc0Tlo = (bf16_t*)alloc((size_t)H_ * E_ * 2);
  bf16_t* WfbThi = (bf16_t*)alloc((size_t)E_ * H_ * 2);
  bf16_t* WfbTlo = (bf16_t*)alloc((size_t)E_ * H_ * 2);
  bf16_t* WihThi = (bf16_t*)alloc((size_t)G4H_ * KIP_ * 2);
  bf16_t* WihTlo = (bf16_t*)alloc((size_t)G4H_ * KIP_ * 2);
  bf16_t* WhhThi = (bf16_t*)alloc((size_t)G4H_ * H_ * 2);
  bf16_t* WhhTlo = (bf16_t*)alloc((size_t)G4H_ * H_ * 2);
  bf16_t* WfcThi = (bf16_t*)alloc((size_t)V_ * H_ * 2);
  bf16_t* WfcTlo = (bf16_t*)alloc((size_t)V_ * H_ * 2);
  float*  meanE  = (float*) alloc((size_t)B_ * E_ * 4);
  bf16_t* meanHi = (bf16_t*)alloc((size_t)B_ * E_ * 2);
  bf16_t* meanLo = (bf16_t*)alloc((size_t)B_ * E_ * 2);
  float*  h      = (float*) alloc((size_t)B_ * H_ * 4);
  float*  c      = (float*) alloc((size_t)B_ * H_ * 4);
  bf16_t* hHi    = (bf16_t*)alloc((size_t)B_ * H_ * 2);
  bf16_t* hLo    = (bf16_t*)alloc((size_t)B_ * H_ * 2);
  float*  attn2  = (float*) alloc((size_t)B_ * A_ * 4);
  float*  logits = (float*) alloc(MP * 4);
  float*  alpha  = (float*) alloc(MP * 4);
  float*  gateL  = (float*) alloc((size_t)B_ * E_ * 4);
  float*  awe    = (float*) alloc((size_t)B_ * E_ * 4);
  bf16_t* inpHi  = (bf16_t*)alloc((size_t)B_ * KIP_ * 2);
  bf16_t* inpLo  = (bf16_t*)alloc((size_t)B_ * KIP_ * 2);
  float*  gates  = (float*) alloc((size_t)B_ * G4H_ * 4);
  if (off > ws_size) return;   // workspace too small (needs ~195 MB)

  // ---- weight prep: transpose + bf16 hi/lo split (once per launch) --------
  auto tr = [&](const float* W, bf16_t* Thi, bf16_t* Tlo, int K, int N, int Kp) {
    int n = N * Kp;
    k_transpose_split<<<(n + 255) / 256, 256, 0, stream>>>(W, Thi, Tlo, K, N, Kp);
  };
  tr(We_a, WeaThi, WeaTlo, E_, A_, E_);
  tr(Wd_a, WdaThi, WdaTlo, H_, A_, H_);
  tr(Wh0,  Wh0Thi, Wh0Tlo, E_, H_, E_);
  tr(Wc0,  Wc0Thi, Wc0Tlo, E_, H_, E_);
  tr(Wfb,  WfbThi, WfbTlo, H_, E_, H_);
  tr(Wih,  WihThi, WihTlo, KI_, G4H_, KIP_);
  tr(Whh,  WhhThi, WhhTlo, H_, G4H_, H_);
  tr(Wfc,  WfcThi, WfcTlo, H_, V_, H_);

  // ---- activations prep ----------------------------------------------------
  {
    long n = (long)MP * E_;
    k_split<<<(int)((n + 255) / 256), 256, 0, stream>>>(x, Xhi, Xlo, n);
  }
  k_mean<<<dim3(E_ / 256, B_), 256, 0, stream>>>(x, meanE, meanHi, meanLo);

  // attn1 = X @ We_a + be_a       [12544 x 512], K = 2048
  gemm(stream, Xhi, Xlo, E_, WeaThi, WeaTlo, E_, be_a, attn1, A_, B_ * P_, A_, E_, 0);
  // h0 / c0
  gemm(stream, meanHi, meanLo, E_, Wh0Thi, Wh0Tlo, E_, bh0, h, H_, B_, H_, E_, 0);
  gemm(stream, meanHi, meanLo, E_, Wc0Thi, Wc0Tlo, E_, bc0, c, H_, B_, H_, E_, 0);
  {
    long n = (long)B_ * H_;
    k_split<<<(int)((n + 255) / 256), 256, 0, stream>>>(h, hHi, hLo, n);
  }

  // ---- sequential decode loop ---------------------------------------------
  for (int t = 0; t < T_; ++t) {
    // attn2 = h @ Wd_a + bd_a
    gemm(stream, hHi, hLo, H_, WdaThi, WdaTlo, H_, bd_a, attn2, A_, B_, A_, H_, 0);
    // logits, softmax (also writes alphas output), awe
    k_attn_logits<<<(B_ * P_) / 8, 256, 0, stream>>>(attn1, attn2, Wf_a, bf_a, logits);
    k_softmax<<<B_, 256, 0, stream>>>(logits, alpha, alphas_out, t);
    k_awe<<<dim3(E_ / 256, B_), 256, 0, stream>>>(x, alpha, awe);
    // gate = sigmoid(h @ Wfb + bfb)  (sigmoid applied in k_build_inp)
    gemm(stream, hHi, hLo, H_, WfbThi, WfbTlo, H_, bfb, gateL, E_, B_, E_, H_, 0);
    // inp = [emb_t, gate*awe, 0-pad] -> bf16 hi/lo
    k_build_inp<<<(B_ * KIP_ + 255) / 256, 256, 0, stream>>>(
        caps, emb, gateL, awe, inpHi, inpLo, t);
    // gates = inp @ Wih + bih + h @ Whh + bhh
    gemm(stream, inpHi, inpLo, KIP_, WihThi, WihTlo, KIP_, bih, gates, G4H_,
         B_, G4H_, KIP_, 0);
    gemm(stream, hHi, hLo, H_, WhhThi, WhhTlo, H_, bhh, gates, G4H_,
         B_, G4H_, H_, 1);
    // LSTM pointwise (updates h, c, h hi/lo planes)
    k_lstm<<<(B_ * H_ + 255) / 256, 256, 0, stream>>>(gates, h, c, hHi, hLo);
    // preds_t = h_new @ Wfc + bfc  -> d_out[b, t, :]
    gemm(stream, hHi, hLo, H_, WfcThi, WfcTlo, H_, bfc,
         preds_out + (size_t)t * V_, (long)T_ * V_, B_, V_, H_, 0);
  }
}
